// Generator_38319698215333
// MI455X (gfx1250) — matrix operand order repaired and checked by reading, not compiled, not run on hardware
//
#include <hip/hip_runtime.h>
#include <hip/hip_bf16.h>
#include <cstddef>

#define BB 8
#define NN 2048
#define KSAMP 32

typedef __attribute__((ext_vector_type(16))) _Float16 v16h;
typedef __attribute__((ext_vector_type(8)))  _Float16 v8h;
typedef __attribute__((ext_vector_type(8)))  float    v8f;

// ---------------- static device workspace (.bss, fully overwritten each call) ----------
#define BUF_ELEMS (33554432u)              /* BB * 64 * 65536 floats = 128 MiB */
#define XT_PER_B  (3276800u)               /* >= max cols*Cpad halves per batch (32768*96) */
__device__ float g_pts0[BB * NN * 3];
__device__ float g_pts1[BB * 2048 * 3];
__device__ float g_pts2[BB * 1024 * 3];
__device__ float g_pts3[BB * 512 * 3];
__device__ float g_pts4[BB * 256 * 3];
__device__ int   g_fidx[BB * 2048];
__device__ int   g_gidx[BB * 2048 * KSAMP];
__device__ float g_feat1[BB * 64 * 2048];
__device__ float g_feat2[BB * 128 * 1024];
__device__ float g_feat3[BB * 256 * 512];
__device__ float g_feat4[BB * 512 * 256];
__device__ float g_up4[BB * 64 * NN];
__device__ float g_up3[BB * 64 * NN];
__device__ float g_up2[BB * 64 * NN];
__device__ float g_concat[BB * 259 * NN];
__device__ float g_net[BB * 552 * NN];
__device__ int   g_ki[BB * NN * 3];
__device__ float g_kw[BB * NN * 3];
__device__ float g_mean[1024];
__device__ float g_rstd[1024];
__device__ float g_bufA[BUF_ELEMS];
__device__ float g_bufB[BUF_ELEMS];
__device__ __attribute__((aligned(256))) _Float16 g_wh[512 * 608];       // padded f16 weights
__device__ __attribute__((aligned(256))) _Float16 g_xt[(size_t)BB * XT_PER_B]; // transposed f16 X

// ---------------- kernels ----------------

// (B,3,N) -> (B,N,3)
__global__ void k_transpose_xyz(const float* __restrict__ src, float* __restrict__ dst) {
  long idx = (long)blockIdx.x * blockDim.x + threadIdx.x;
  if (idx >= (long)BB * NN * 3) return;
  int c = (int)(idx % 3);
  int n = (int)((idx / 3) % NN);
  int b = (int)(idx / (3L * NN));
  dst[idx] = src[((size_t)b * 3 + c) * NN + n];
}

// Farthest-point sampling: one block per batch. pts (B,Np,3), fidx (B,npoint).
__global__ void __launch_bounds__(256) k_fps(const float* __restrict__ pts, int Np, int npoint,
                                             int* __restrict__ fidx) {
  int b = blockIdx.x;
  const float* P = pts + (size_t)b * Np * 3;
  __shared__ float spx[2048], spy[2048], spz[2048], dist[2048];
  __shared__ float rv[256];
  __shared__ int   ri[256];
  __shared__ int   s_far;
  for (int j = threadIdx.x; j < Np; j += 256) {
    spx[j] = P[j * 3 + 0]; spy[j] = P[j * 3 + 1]; spz[j] = P[j * 3 + 2];
    dist[j] = 1e10f;
  }
  if (threadIdx.x == 0) s_far = 0;
  __syncthreads();
  for (int i = 0; i < npoint; ++i) {
    int far = s_far;
    if (threadIdx.x == 0) fidx[b * npoint + i] = far;
    float cx = spx[far], cy = spy[far], cz = spz[far];
    float best = -1.0f; int bi = 0;
    for (int j = threadIdx.x; j < Np; j += 256) {
      float dx = spx[j] - cx, dy = spy[j] - cy, dz = spz[j] - cz;
      float d = dx * dx + dy * dy + dz * dz;
      float dd = fminf(dist[j], d);
      dist[j] = dd;
      if (dd > best) { best = dd; bi = j; }
    }
    rv[threadIdx.x] = best; ri[threadIdx.x] = bi;
    __syncthreads();
    for (int off = 128; off > 0; off >>= 1) {
      if (threadIdx.x < off) {
        float ov = rv[threadIdx.x + off]; int oi = ri[threadIdx.x + off];
        if (ov > rv[threadIdx.x] || (ov == rv[threadIdx.x] && oi < ri[threadIdx.x])) {
          rv[threadIdx.x] = ov; ri[threadIdx.x] = oi;
        }
      }
      __syncthreads();
    }
    if (threadIdx.x == 0) s_far = ri[0];
    __syncthreads();
  }
}

// dst(B,S,3) = src(B,Np,3)[fidx]
__global__ void k_gather_pts(const float* __restrict__ src, const int* __restrict__ fidx,
                             int Np, int S, float* __restrict__ dst) {
  long idx = (long)blockIdx.x * blockDim.x + threadIdx.x;
  if (idx >= (long)BB * S * 3) return;
  int c = (int)(idx % 3);
  int s = (int)((idx / 3) % S);
  int b = (int)(idx / (3L * S));
  int g = fidx[b * S + s];
  dst[idx] = src[((size_t)b * Np + g) * 3 + c];
}

// Ball query: first KSAMP indices with squared dist <= r2, padded with first hit.
__global__ void k_query_ball(const float* __restrict__ xyz, const float* __restrict__ nxyz,
                             int Np, int S, float r2, int* __restrict__ gidx) {
  int t = blockIdx.x * blockDim.x + threadIdx.x;
  if (t >= BB * S) return;
  int b = t / S, s = t % S;
  (void)s;
  const float* P = xyz + (size_t)b * Np * 3;
  float qx = nxyz[(size_t)t * 3 + 0], qy = nxyz[(size_t)t * 3 + 1], qz = nxyz[(size_t)t * 3 + 2];
  int* g = gidx + (size_t)t * KSAMP;
  int cnt = 0;
  for (int j = 0; j < Np && cnt < KSAMP; ++j) {
    float dx = P[j * 3 + 0] - qx, dy = P[j * 3 + 1] - qy, dz = P[j * 3 + 2] - qz;
    if (dx * dx + dy * dy + dz * dz <= r2) g[cnt++] = j;
  }
  int f = (cnt > 0) ? g[0] : 0;
  for (; cnt < KSAMP; ++cnt) g[cnt] = f;
}

// Build grouped input Xg (B, 3+Cf, S*KSAMP); col = s*KSAMP + k.
__global__ void k_group(const float* __restrict__ xyz, const float* __restrict__ nxyz,
                        const float* __restrict__ feat, const int* __restrict__ gidx,
                        int Np, int S, int Cf, float* __restrict__ Xg, long total) {
  long idx = (long)blockIdx.x * blockDim.x + threadIdx.x;
  if (idx >= total) return;
  int C = 3 + Cf;
  long cols = (long)S * KSAMP;
  int col = (int)(idx % cols);
  int c = (int)((idx / cols) % C);
  int b = (int)(idx / (cols * C));
  int s = col / KSAMP, k = col % KSAMP;
  int g = gidx[((size_t)b * S + s) * KSAMP + k];
  float v;
  if (c < 3) v = xyz[((size_t)b * Np + g) * 3 + c] - nxyz[((size_t)b * S + s) * 3 + c];
  else       v = feat[((size_t)b * Cf + (c - 3)) * Np + g];
  Xg[idx] = v;
}

// W (OxC f32) -> Wh (Opad x Cpad f16, zero padded)
__global__ void k_cvt_w(const float* __restrict__ W, _Float16* __restrict__ Wh,
                        int O, int C, int Opad, int Cpad) {
  int idx = blockIdx.x * blockDim.x + threadIdx.x;
  if (idx >= Opad * Cpad) return;
  int k = idx % Cpad, m = idx / Cpad;
  float v = (m < O && k < C) ? W[(size_t)m * C + k] : 0.0f;
  Wh[idx] = (_Float16)v;
}

// X (B,C,cols f32) -> XT (B, cols, Cpad f16), transposed + zero padded
__global__ void k_cvt_x(const float* __restrict__ X, _Float16* __restrict__ XT,
                        int C, int Cpad, int cols, long total) {
  long idx = (long)blockIdx.x * blockDim.x + threadIdx.x;
  if (idx >= total) return;
  int c = (int)(idx % Cpad);
  int s = (int)((idx / Cpad) % cols);
  int b = (int)(idx / ((long)Cpad * cols));
  float v = (c < C) ? X[((size_t)b * C + c) * cols + s] : 0.0f;
  XT[((size_t)b * cols + s) * (size_t)Cpad + c] = (_Float16)v;
}

// WMMA GEMM: Y[b] = W(OxC) * X[b](C x cols) + bias, from pre-formatted f16 operands.
// One wave computes a 16x16 tile; 4 waves per block along the column axis.
// No guards inside the k-loop: operands are zero-padded, all loads are aligned b128.
__global__ void __launch_bounds__(128) k_gemm_wmma(const _Float16* __restrict__ Wh,
                                                   const float* __restrict__ bias,
                                                   const _Float16* __restrict__ XT,
                                                   float* __restrict__ Y,
                                                   int O, int Cpad, int cols) {
  int wave = threadIdx.x >> 5;
  int lane = threadIdx.x & 31;
  int s0 = (blockIdx.x * 4 + wave) * 16;
  int o0 = blockIdx.y * 16;
  int b  = blockIdx.z;
  if (s0 >= cols) return;                    // wave-uniform
  int hi = lane >> 4;                        // lane half (0/1)
  int lr = lane & 15;
  int m  = o0 + lr;                          // < Opad by construction
  int s  = s0 + lr;                          // cols is always a multiple of 16
  const _Float16* Wrow = Wh + (size_t)m * Cpad;
  const _Float16* Xrow = XT + ((size_t)b * cols + s) * (size_t)Cpad;
  v8f acc = {};
  for (int c0 = 0; c0 < Cpad; c0 += 32) {
    // A fragment: e=0..7 -> K=c0+hi*8+e ; e=8..15 -> K=c0+16+hi*8+(e-8)
    v8h a_lo = *(const v8h*)(Wrow + c0 + hi * 8);
    v8h a_hi = *(const v8h*)(Wrow + c0 + 16 + hi * 8);
    // B fragment: e=0..7 -> K=c0+hi*8+e ; e=8..15 -> K=c0+16+hi*8+(e-8)
    v8h b_lo = *(const v8h*)(Xrow + c0 + hi * 8);
    v8h b_hi = *(const v8h*)(Xrow + c0 + 16 + hi * 8);
    v16h av = __builtin_shufflevector(a_lo, a_hi, 0, 1, 2, 3, 4, 5, 6, 7,
                                      8, 9, 10, 11, 12, 13, 14, 15);
    v16h bv = __builtin_shufflevector(b_lo, b_hi, 0, 1, 2, 3, 4, 5, 6, 7,
                                      8, 9, 10, 11, 12, 13, 14, 15);
    acc = __builtin_amdgcn_wmma_f32_16x16x32_f16(false, av, false, bv,
                                                 (short)0, acc, false, false);
  }
  float* Yb = Y + (size_t)b * O * cols;
  #pragma unroll
  for (int r = 0; r < 8; ++r) {              // D: VGPR r -> M = r + 8*hi, N = lane%16
    int mm = o0 + r + (hi ? 8 : 0);
    if (mm < O) Yb[(size_t)mm * cols + s] = acc[r] + bias[mm];
  }
}

// Per-channel batch statistics over (B, cols). One block per channel.
__global__ void __launch_bounds__(256) k_bn_stats(const float* __restrict__ Y, int O, long cols,
                                                  float* __restrict__ mean, float* __restrict__ rstd) {
  int o = blockIdx.x;
  __shared__ float ss[256], sq[256];
  float s = 0.f, q = 0.f;
  for (int b = 0; b < BB; ++b) {
    const float* p = Y + ((size_t)b * O + o) * cols;
    for (long j = threadIdx.x; j < cols; j += 256) { float v = p[j]; s += v; q += v * v; }
  }
  ss[threadIdx.x] = s; sq[threadIdx.x] = q;
  __syncthreads();
  for (int off = 128; off > 0; off >>= 1) {
    if (threadIdx.x < off) { ss[threadIdx.x] += ss[threadIdx.x + off]; sq[threadIdx.x] += sq[threadIdx.x + off]; }
    __syncthreads();
  }
  if (threadIdx.x == 0) {
    float n = (float)((long)BB * cols);
    float mval = ss[0] / n;
    float var  = sq[0] / n - mval * mval;
    mean[o] = mval;
    rstd[o] = rsqrtf(fmaxf(var, 0.f) + 1e-5f);
  }
}

__global__ void k_bn_relu(float* __restrict__ Y, const float* __restrict__ mean,
                          const float* __restrict__ rstd, const float* __restrict__ gamma,
                          const float* __restrict__ beta, int O, long cols, long total) {
  long idx = (long)blockIdx.x * blockDim.x + threadIdx.x;
  if (idx >= total) return;
  int o = (int)((idx / cols) % O);
  float v = Y[idx];
  v = gamma[o] * (v - mean[o]) * rstd[o] + beta[o];
  Y[idx] = fmaxf(v, 0.0f);
}

// out(B,O,S) = max over k of Y(B,O,S*KSAMP) with col = s*KSAMP+k
__global__ void k_maxpool(const float* __restrict__ Y, float* __restrict__ out, int O, int S) {
  long idx = (long)blockIdx.x * blockDim.x + threadIdx.x;
  if (idx >= (long)BB * O * S) return;
  int s = (int)(idx % S);
  int o = (int)((idx / S) % O);
  int b = (int)(idx / ((long)S * O));
  const float* p = Y + ((size_t)b * O + o) * ((size_t)S * KSAMP) + (size_t)s * KSAMP;
  float m = p[0];
  #pragma unroll 4
  for (int k = 1; k < KSAMP; ++k) m = fmaxf(m, p[k]);
  out[idx] = m;
}

// 3-NN (squared distances) + inverse-distance weights
__global__ void k_knn3(const float* __restrict__ q, const float* __restrict__ p, int S,
                       int* __restrict__ ki, float* __restrict__ kw) {
  int t = blockIdx.x * blockDim.x + threadIdx.x;
  if (t >= BB * NN) return;
  int b = t / NN;
  float qx = q[(size_t)t * 3 + 0], qy = q[(size_t)t * 3 + 1], qz = q[(size_t)t * 3 + 2];
  const float* P = p + (size_t)b * S * 3;
  float d0 = 3.4e38f, d1 = 3.4e38f, d2 = 3.4e38f;
  int i0 = 0, i1 = 0, i2 = 0;
  for (int j = 0; j < S; ++j) {
    float dx = P[j * 3 + 0] - qx, dy = P[j * 3 + 1] - qy, dz = P[j * 3 + 2] - qz;
    float d = dx * dx + dy * dy + dz * dz;
    if (d < d0)      { d2 = d1; i2 = i1; d1 = d0; i1 = i0; d0 = d; i0 = j; }
    else if (d < d1) { d2 = d1; i2 = i1; d1 = d;  i1 = j; }
    else if (d < d2) { d2 = d;  i2 = j; }
  }
  float r0 = 1.f / (d0 + 1e-8f), r1 = 1.f / (d1 + 1e-8f), r2 = 1.f / (d2 + 1e-8f);
  float sum = r0 + r1 + r2;
  ki[t * 3 + 0] = i0; ki[t * 3 + 1] = i1; ki[t * 3 + 2] = i2;
  kw[t * 3 + 0] = r0 / sum; kw[t * 3 + 1] = r1 / sum; kw[t * 3 + 2] = r2 / sum;
}

// interp(B,C,NN) from feat(B,C,S)
__global__ void k_interp(const float* __restrict__ feat, const int* __restrict__ ki,
                         const float* __restrict__ kw, int C, int S, float* __restrict__ out,
                         long total) {
  long idx = (long)blockIdx.x * blockDim.x + threadIdx.x;
  if (idx >= total) return;
  int n = (int)(idx % NN);
  int c = (int)((idx / NN) % C);
  int b = (int)(idx / ((long)NN * C));
  const float* F = feat + ((size_t)b * C + c) * S;
  long base = ((long)b * NN + n) * 3;
  out[idx] = kw[base + 0] * F[ki[base + 0]] +
             kw[base + 1] * F[ki[base + 1]] +
             kw[base + 2] * F[ki[base + 2]];
}

__global__ void k_copy_channels(float* __restrict__ dst, int Cd, int coff,
                                const float* __restrict__ src, int Cs, int cols, long total) {
  long idx = (long)blockIdx.x * blockDim.x + threadIdx.x;
  if (idx >= total) return;
  int j = (int)(idx % cols);
  int c = (int)((idx / cols) % Cs);
  int b = (int)(idx / ((long)cols * Cs));
  dst[((size_t)b * Cd + coff + c) * cols + j] = src[idx];
}

__global__ void k_labels(float* __restrict__ net, const float* __restrict__ lab, long total) {
  long idx = (long)blockIdx.x * blockDim.x + threadIdx.x;
  if (idx >= total) return;
  int n = (int)(idx % NN);
  int c = (int)((idx / NN) % 40);
  int b = (int)(idx / (40L * NN));
  net[((size_t)b * 552 + 512 + c) * NN + n] = lab[b * 40 + c];
}

__global__ void k_copyf(float* __restrict__ dst, const float* __restrict__ src, long n) {
  long i = (long)blockIdx.x * blockDim.x + threadIdx.x;
  if (i < n) dst[i] = src[i];
}

// ---------------- host orchestration ----------------

static inline unsigned grd(long total) { return (unsigned)((total + 255) / 256); }

#define GETSYM(var, symname, type)                                   \
  type* var = nullptr;                                               \
  (void)hipGetSymbolAddress((void**)&var, HIP_SYMBOL(symname));

extern "C" void kernel_launch(void* const* d_in, const int* in_sizes, int n_in,
                              void* d_out, int out_size, void* d_ws, size_t ws_size,
                              hipStream_t stream) {
  (void)in_sizes; (void)n_in; (void)d_ws; (void)ws_size; (void)out_size;
  const float* xyz_in = (const float*)d_in[0];
  const float* lab_in = (const float*)d_in[1];
  auto IN = [&](int i) { return (const float*)d_in[i]; };

  GETSYM(pts0, g_pts0, float)   GETSYM(pts1, g_pts1, float)
  GETSYM(pts2, g_pts2, float)   GETSYM(pts3, g_pts3, float)
  GETSYM(pts4, g_pts4, float)
  GETSYM(fidxp, g_fidx, int)    GETSYM(gidxp, g_gidx, int)
  GETSYM(feat1, g_feat1, float) GETSYM(feat2, g_feat2, float)
  GETSYM(feat3, g_feat3, float) GETSYM(feat4, g_feat4, float)
  GETSYM(up4, g_up4, float)     GETSYM(up3, g_up3, float)   GETSYM(up2, g_up2, float)
  GETSYM(concatp, g_concat, float) GETSYM(netp, g_net, float)
  GETSYM(kip, g_ki, int)        GETSYM(kwp, g_kw, float)
  GETSYM(meanp, g_mean, float)  GETSYM(rstdp, g_rstd, float)
  GETSYM(bufA, g_bufA, float)   GETSYM(bufB, g_bufB, float)
  GETSYM(whp, g_wh, _Float16)   GETSYM(xtp, g_xt, _Float16)

  // GEMM wrapper: stage f16 operands (padded / transposed), then WMMA.
  auto gemm = [&](const float* W, const float* bias, const float* X, float* Y,
                  int O, int C, int cols) {
    int Cpad = (C + 31) & ~31;
    int Opad = (O + 15) & ~15;
    k_cvt_w<<<grd((long)Opad * Cpad), 256, 0, stream>>>(W, whp, O, C, Opad, Cpad);
    long tx = (long)BB * cols * Cpad;
    k_cvt_x<<<grd(tx), 256, 0, stream>>>(X, xtp, C, Cpad, cols, tx);
    dim3 grid((unsigned)((cols + 63) / 64), (unsigned)(Opad / 16), BB);
    k_gemm_wmma<<<grid, 128, 0, stream>>>(whp, bias, xtp, Y, O, Cpad, cols);
  };

  auto bn = [&](float* Y, const float* gamma, const float* beta, int O, long cols) {
    k_bn_stats<<<O, 256, 0, stream>>>(Y, O, cols, meanp, rstdp);
    long total = (long)BB * O * cols;
    k_bn_relu<<<grd(total), 256, 0, stream>>>(Y, meanp, rstdp, gamma, beta, O, cols, total);
  };

  auto ccopy = [&](float* dst, int Cd, int coff, const float* src, int Cs, int cols) {
    long t = (long)BB * Cs * cols;
    k_copy_channels<<<grd(t), 256, 0, stream>>>(dst, Cd, coff, src, Cs, cols, t);
  };

  // (B,3,N) -> (B,N,3)
  k_transpose_xyz<<<grd((long)BB * NN * 3), 256, 0, stream>>>(xyz_in, pts0);

  auto sa_level = [&](const float* src_pts, int Np, const float* feat, int Cf, int S, float r2,
                      int pb, int O1, int O2, int O3, float* dst_pts, float* dst_feat) {
    int C = 3 + Cf;
    long cols = (long)S * KSAMP;
    k_fps<<<BB, 256, 0, stream>>>(src_pts, Np, S, fidxp);
    k_gather_pts<<<grd((long)BB * S * 3), 256, 0, stream>>>(src_pts, fidxp, Np, S, dst_pts);
    k_query_ball<<<grd((long)BB * S), 256, 0, stream>>>(src_pts, dst_pts, Np, S, r2, gidxp);
    long tg = (long)BB * C * cols;
    k_group<<<grd(tg), 256, 0, stream>>>(src_pts, dst_pts, feat, gidxp, Np, S, Cf, bufA, tg);
    gemm(IN(pb + 0), IN(pb + 1), bufA, bufB, O1, C, (int)cols);
    bn(bufB, IN(pb + 2), IN(pb + 3), O1, cols);
    gemm(IN(pb + 4), IN(pb + 5), bufB, bufA, O2, O1, (int)cols);
    bn(bufA, IN(pb + 6), IN(pb + 7), O2, cols);
    gemm(IN(pb + 8), IN(pb + 9), bufA, bufB, O3, O2, (int)cols);
    bn(bufB, IN(pb + 10), IN(pb + 11), O3, cols);
    k_maxpool<<<grd((long)BB * O3 * S), 256, 0, stream>>>(bufB, dst_feat, O3, S);
  };

  sa_level(pts0, NN,   nullptr, 0,   2048, 0.05f * 0.05f,  2,  32,  32,  64, pts1, feat1);
  sa_level(pts1, 2048, feat1,   64,  1024, 0.10f * 0.10f, 14,  64,  64, 128, pts2, feat2);
  sa_level(pts2, 1024, feat2,   128,  512, 0.20f * 0.20f, 26, 128, 128, 256, pts3, feat3);
  sa_level(pts3, 512,  feat3,   256,  256, 0.30f * 0.30f, 38, 256, 256, 512, pts4, feat4);

  auto fprop = [&](const float* srcp, int S, const float* featv, int Cf, int pb, float* dst) {
    k_knn3<<<grd((long)BB * NN), 256, 0, stream>>>(pts0, srcp, S, kip, kwp);
    long ti = (long)BB * Cf * NN;
    k_interp<<<grd(ti), 256, 0, stream>>>(featv, kip, kwp, Cf, S, bufA, ti);
    gemm(IN(pb), IN(pb + 1), bufA, bufB, 64, Cf, NN);
    bn(bufB, IN(pb + 2), IN(pb + 3), 64, NN);
    long tc = (long)BB * 64 * NN;
    k_copyf<<<grd(tc), 256, 0, stream>>>(dst, bufB, tc);
  };

  fprop(pts4, 256,  feat4, 512, 50, up4);
  fprop(pts3, 512,  feat3, 256, 54, up3);
  fprop(pts2, 1024, feat2, 128, 58, up2);

  // concat: [up4, up3, up2, l1_points, xyz] -> (B,259,N)
  ccopy(concatp, 259, 0,   up4,    64, NN);
  ccopy(concatp, 259, 64,  up3,    64, NN);
  ccopy(concatp, 259, 128, up2,    64, NN);
  ccopy(concatp, 259, 192, feat1,  64, NN);
  ccopy(concatp, 259, 256, xyz_in, 3,  NN);

  // four up-MLPs (259 -> 256 -> 128) into net channels [u*128, u*128+128)
  for (int u = 0; u < 4; ++u) {
    int pb = 62 + u * 8;
    gemm(IN(pb + 0), IN(pb + 1), concatp, bufA, 256, 259, NN);
    bn(bufA, IN(pb + 2), IN(pb + 3), 256, NN);
    gemm(IN(pb + 4), IN(pb + 5), bufA, bufB, 128, 256, NN);
    bn(bufB, IN(pb + 6), IN(pb + 7), 128, NN);
    ccopy(netp, 552, u * 128, bufB, 128, NN);
  }

  // labels broadcast into channels 512..551
  k_labels<<<grd((long)BB * 40 * NN), 256, 0, stream>>>(netp, lab_in, (long)BB * 40 * NN);

  // head: 552 -> 64 -> 3 (BN+ReLU on both layers, matching reference)
  gemm(IN(94), IN(95), netp, bufA, 64, 552, NN);
  bn(bufA, IN(96), IN(97), 64, NN);
  gemm(IN(98), IN(99), bufA, bufB, 3, 64, NN);
  bn(bufB, IN(100), IN(101), 3, NN);

  long to = (long)BB * 3 * NN;
  k_copyf<<<grd(to), 256, 0, stream>>>((float*)d_out, bufB, to);
}
